// ECF_73796128079974
// MI455X (gfx1250) — matrix operand order repaired and checked by reading, not compiled, not run on hardware
//
#include <hip/hip_runtime.h>

typedef _Float16 h16;
typedef __attribute__((ext_vector_type(16))) _Float16 v16h;
typedef __attribute__((ext_vector_type(8)))  float    v8f;
typedef __attribute__((ext_vector_type(4)))  unsigned u32x4;
typedef __attribute__((ext_vector_type(2)))  unsigned u32x2;

#define B_   16
#define C_   256
#define H_   64
#define W_   64
#define HW_  4096
#define CHW_ (C_*HW_)
#define OUT_F 16777216   // B*C*H*W

// ---------------- stats: gr/gd = mean over (H,W) per (b,c) ----------------
__global__ void __launch_bounds__(256) k_stats(const float* __restrict__ f1,
                                               const float* __restrict__ f2,
                                               float* __restrict__ gr,
                                               float* __restrict__ gd) {
  __shared__ float r1[256], r2[256];
  const int t = threadIdx.x;
  const size_t base = (size_t)blockIdx.x * HW_;
  float s1 = 0.f, s2 = 0.f;
  for (int i = t; i < HW_; i += 256) { s1 += f1[base + i]; s2 += f2[base + i]; }
  r1[t] = s1; r2[t] = s2; __syncthreads();
  for (int s = 128; s > 0; s >>= 1) {
    if (t < s) { r1[t] += r1[t + s]; r2[t] += r2[t + s]; }
    __syncthreads();
  }
  if (t == 0) { gr[blockIdx.x] = r1[0] * (1.f / HW_); gd[blockIdx.x] = r2[0] * (1.f / HW_); }
}

// ---------------- channel pooling: mean_c + max_c ----------------
__global__ void __launch_bounds__(256) k_pool(const float* __restrict__ f1,
                                              const float* __restrict__ f2,
                                              float* __restrict__ f1in,
                                              float* __restrict__ f2in) {
  const int gid = blockIdx.x * 256 + threadIdx.x;   // 0..65535 = b*4096+hw
  const int b = gid >> 12, hw = gid & 4095;
  const size_t base = (size_t)b * CHW_ + hw;
  float s1 = 0.f, s2 = 0.f, m1 = -3.4e38f, m2 = -3.4e38f;
  for (int c = 0; c < C_; ++c) {
    float v1 = f1[base + (size_t)c * HW_];
    float v2 = f2[base + (size_t)c * HW_];
    s1 += v1; s2 += v2; m1 = fmaxf(m1, v1); m2 = fmaxf(m2, v2);
  }
  f1in[gid] = s1 * (1.f / C_) + m1;
  f2in[gid] = s2 * (1.f / C_) + m2;
}

// ---------------- MCA MLPs + 2-way softmax ----------------
__global__ void __launch_bounds__(256) k_mlp(
    const float* __restrict__ gr, const float* __restrict__ gd,
    const float* __restrict__ w1r, const float* __restrict__ b1r,
    const float* __restrict__ w2r, const float* __restrict__ b2r,
    const float* __restrict__ w1d, const float* __restrict__ b1d,
    const float* __restrict__ w2d, const float* __restrict__ b2d,
    float* __restrict__ wrw, float* __restrict__ wdw, float* __restrict__ out) {
  __shared__ float grb[256], gdb[256], hr[16], hd[16];
  const int t = threadIdx.x, b = blockIdx.x;
  grb[t] = gr[b * 256 + t]; gdb[t] = gd[b * 256 + t];
  __syncthreads();
  if (t < 16) {
    float a = b1r[t];
    for (int i = 0; i < 256; ++i) a += grb[i] * w1r[t * 256 + i];
    hr[t] = fmaxf(a, 0.f);
  } else if (t < 32) {
    const int j = t - 16;
    float a = b1d[j];
    for (int i = 0; i < 256; ++i) a += gdb[i] * w1d[j * 256 + i];
    hd[j] = fmaxf(a, 0.f);
  }
  __syncthreads();
  float lr = b2r[t], ld = b2d[t];
  for (int j = 0; j < 16; ++j) { lr += hr[j] * w2r[t * 16 + j]; ld += hd[j] * w2d[t * 16 + j]; }
  const float mx = fmaxf(lr, ld);
  const float er = __expf(lr - mx), ed = __expf(ld - mx);
  const float inv = 1.f / (er + ed);
  const float wrv = er * inv, wdv = ed * inv;
  wrw[b * 256 + t] = wrv; wdw[b * 256 + t] = wdv;
  out[OUT_F + b * 256 + t] = wrv;
  out[OUT_F + 4096 + b * 256 + t] = wdv;
}

// ---------------- per-batch spatial attention (both branches) ----------------
__device__ void attn_branch(const float* mq, const float* mv,
                            float qw, float qb, float kw, float kb2,
                            float vw, float vb,
                            float* att, float* red, float* sout, int rg, int t) {
  const int r = t >> 6, k = t & 63;
  const int h = rg * 4 + r;
  float dot = 0.f;
  for (int w = 0; w < 64; ++w)
    dot += (qw * mq[h * 64 + w] + qb) * (kw * mq[k * 64 + w] + kb2);
  red[r * 64 + k] = dot;
  __syncthreads();
  for (int s = 32; s > 0; s >>= 1) {
    if (k < s) red[r * 64 + k] = fmaxf(red[r * 64 + k], red[r * 64 + k + s]);
    __syncthreads();
  }
  const float mx = red[r * 64];
  __syncthreads();
  const float p = __expf(dot - mx);
  red[r * 64 + k] = p;
  __syncthreads();
  for (int s = 32; s > 0; s >>= 1) {
    if (k < s) red[r * 64 + k] += red[r * 64 + k + s];
    __syncthreads();
  }
  const float sm = red[r * 64];
  __syncthreads();
  att[r * 64 + k] = p / sm;
  __syncthreads();
  const int w = k;
  float acc = 0.f;
  for (int kk = 0; kk < 64; ++kk)
    acc += att[r * 64 + kk] * (vw * mv[kk * 64 + w] + vb);
  acc += vw * mv[h * 64 + w] + vb;
  sout[h * 64 + w] = acc;
  __syncthreads();
}

__global__ void __launch_bounds__(256) k_attn(
    const float* __restrict__ f1in, const float* __restrict__ f2in,
    const float* qr_w, const float* qr_b, const float* kr_w, const float* kr_b,
    const float* vr_w, const float* vr_b,
    const float* qd_w, const float* qd_b, const float* kd_w, const float* kd_b,
    const float* vd_w, const float* vd_b,
    float* __restrict__ s1, float* __restrict__ s2) {
  __shared__ float m1[4096], m2[4096];
  __shared__ float att[256], red[256];
  const int t = threadIdx.x, rg = blockIdx.x, b = blockIdx.y;
  for (int i = t; i < 4096; i += 256) {
    m1[i] = f1in[(size_t)b * 4096 + i];
    m2[i] = f2in[(size_t)b * 4096 + i];
  }
  __syncthreads();
  attn_branch(m1, m2, *qr_w, *qr_b, *kr_w, *kr_b, *vd_w, *vd_b,
              att, red, s1 + (size_t)b * 4096, rg, t);
  attn_branch(m2, m1, *qd_w, *qd_b, *kd_w, *kd_b, *vr_w, *vr_b,
              att, red, s2 + (size_t)b * 4096, rg, t);
}

// ---------------- fold conv bias + BN into scale/shift ----------------
__global__ void __launch_bounds__(256) k_bnprep(
    const float* bnr_g, const float* bnr_b, const float* bnr_m, const float* bnr_v,
    const float* convr_b,
    const float* bnd_g, const float* bnd_b, const float* bnd_m, const float* bnd_v,
    const float* convd_b,
    float* __restrict__ sc, float* __restrict__ sh) {
  const int br = blockIdx.x, t = threadIdx.x;
  const float* g  = br ? bnd_g : bnr_g;
  const float* bb = br ? bnd_b : bnr_b;
  const float* mm = br ? bnd_m : bnr_m;
  const float* vv = br ? bnd_v : bnr_v;
  const float* cb = br ? convd_b : convr_b;
  const float s = g[t] * rsqrtf(vv[t] + 1e-5f);
  sc[br * 256 + t] = s;
  sh[br * 256 + t] = (cb[t] - mm[t]) * s + bb[t];
}

// ---------------- pack conv weights -> f16 [branch][tap][kb][co][kc32] ----------------
__global__ void __launch_bounds__(256) k_wpack(const float* __restrict__ cwr,
                                               const float* __restrict__ cwd,
                                               h16* __restrict__ wp) {
  const size_t d = (size_t)blockIdx.x * 256 + threadIdx.x;
  if (d >= (size_t)2 * 589824) return;
  const int branch = (int)(d / 589824);
  const int r = (int)(d % 589824);
  const int tapkb = r >> 13;          // 0..71
  const int rem = r & 8191;
  const int co = rem >> 5;
  const int kcl = rem & 31;
  const int tap = tapkb >> 3;
  const int kb = tapkb & 7;
  const int ci = kb * 32 + kcl;
  const float* src = branch ? cwd : cwr;
  wp[d] = (h16)src[(size_t)co * 2304 + ci * 9 + tap];
}

// ---------------- stage activations: x = f*weight + s, NHWC f16 ----------------
__global__ void __launch_bounds__(256) k_pack(
    const float* __restrict__ f1, const float* __restrict__ f2,
    const float* __restrict__ wr, const float* __restrict__ wd,
    const float* __restrict__ s1, const float* __restrict__ s2,
    h16* __restrict__ xh1, h16* __restrict__ xh2) {
  __shared__ h16 tile[64 * 256];
  __shared__ float wrs[256], wds[256];
  const int t = threadIdx.x;
  const int h = blockIdx.x, b = blockIdx.y;
  wrs[t] = wr[b * 256 + t];
  wds[t] = wd[b * 256 + t];
  const int w = t & 63;
  const int cp = t >> 6;
  const float s1v = s1[(size_t)(b * 64 + h) * 64 + w];
  const float s2v = s2[(size_t)(b * 64 + h) * 64 + w];
  __syncthreads();
  for (int it = 0; it < 64; ++it) {
    const int c = it * 4 + cp;
    const float v = f1[(((size_t)b * 256 + c) * 64 + h) * 64 + w];
    tile[w * 256 + c] = (h16)(v * wrs[c] + s1v);
  }
  __syncthreads();
  for (int it = 0; it < 64; ++it)
    xh1[((size_t)b * 4096 + h * 64 + it) * 256 + t] = tile[it * 256 + t];
  __syncthreads();
  for (int it = 0; it < 64; ++it) {
    const int c = it * 4 + cp;
    const float v = f2[(((size_t)b * 256 + c) * 64 + h) * 64 + w];
    tile[w * 256 + c] = (h16)(v * wds[c] + s2v);
  }
  __syncthreads();
  for (int it = 0; it < 64; ++it)
    xh2[((size_t)b * 4096 + h * 64 + it) * 256 + t] = tile[it * 256 + t];
}

// low 32 bits of an LDS generic address are the LDS byte offset (ISA 10.2)
__device__ __forceinline__ unsigned lds_lo(const void* p) {
  return (unsigned)(size_t)p;
}

// ---------------- fused dual conv3x3 + BN + LeakyReLU + add (WMMA, async DB) ----------------
__global__ void __launch_bounds__(256) k_conv(
    const h16* __restrict__ xh1, const h16* __restrict__ xh2,
    const h16* __restrict__ wp1, const h16* __restrict__ wp2,
    const float* __restrict__ sc, const float* __restrict__ sh,
    float* __restrict__ out) {
  __shared__ __align__(16) h16 As[2][1280];   // 32 M x 32 K, padded stride 40, x2 buffers
  __shared__ __align__(16) h16 Bs[2][8192];   // [co][kc32] 16KB, x2 buffers
  __shared__ float Os[32 * 256];

  const int t = threadIdx.x;
  const int lane = t & 31;
  const int wv = t >> 5;          // 0..7
  const int msub = wv & 1;        // which 16-row half of M
  const int nq = wv >> 1;         // 0..3, 64-column slice of N
  const int b = blockIdx.y;
  const int mt = blockIdx.x;      // 0..127
  const int h = mt >> 1;
  const int w0 = (mt & 1) << 5;

  const int g = lane >> 4;        // K half selector per ISA 16-bit striping
  const int n15 = lane & 15;
  const int mrow = msub * 16 + n15;

  const int am = t >> 3;          // A-tile fill: m row 0..31
  const int ak = (t & 7) << 2;    // A-tile fill: 4 halves at kc

  // A tile fill: global (sync) load + ds_store with halo zero-fill
  auto issueA = [&](const h16* __restrict__ xh, int tile, int buf) {
    const int tap = tile >> 3, kb = tile & 7;
    const int dh = tap / 3 - 1, dw = tap % 3 - 1;
    const int hh = h + dh;
    const int ww = w0 + am + dw;
    const bool aok = (hh >= 0) && (hh < H_) && (ww >= 0) && (ww < W_);
    u32x2 av = {0u, 0u};
    if (aok)
      av = *(const u32x2*)(xh + (((size_t)b * H_ + hh) * W_ + ww) * C_ + kb * 32 + ak);
    *(u32x2*)(&As[buf][am * 40 + ak]) = av;
  };

  // B tile fill: 16KB contiguous, async memory->LDS (4 x b128 per lane)
  auto issueB = [&](const h16* __restrict__ wp, int tile, int buf) {
    const unsigned lb = lds_lo(&Bs[buf][0]) + (unsigned)(t * 16);
    const unsigned voff = (unsigned)(tile * 16384 + t * 16);
    asm volatile(
        "global_load_async_to_lds_b128 %0, %1, %2 offset:0\n\t"
        "global_load_async_to_lds_b128 %0, %1, %2 offset:4096\n\t"
        "global_load_async_to_lds_b128 %0, %1, %2 offset:8192\n\t"
        "global_load_async_to_lds_b128 %0, %1, %2 offset:12288"
        :: "v"(lb), "v"(voff), "s"(wp) : "memory");
  };

  float sum[4][8];
#pragma unroll
  for (int j = 0; j < 4; ++j)
#pragma unroll
    for (int r = 0; r < 8; ++r) sum[j][r] = 0.f;

  for (int br = 0; br < 2; ++br) {
    const h16* __restrict__ xh = br ? xh2 : xh1;
    const h16* __restrict__ wp = br ? wp2 : wp1;
    v8f z = {};
    v8f acc[4] = {z, z, z, z};

    // pipeline prologue: tiles 0 and 1 in flight
    issueA(xh, 0, 0); issueB(wp, 0, 0);
    issueA(xh, 1, 1); issueB(wp, 1, 1);

    for (int ks = 0; ks < 72; ++ks) {
      const int c = ks & 1;
      // tile ks's 4 async ops are the oldest; allow next tile's 4 to stay in flight
      if (ks < 71)
        asm volatile("s_wait_asynccnt 0x4" ::: "memory");
      else
        asm volatile("s_wait_asynccnt 0x0" ::: "memory");
      __syncthreads();   // tile ks visible to all waves (also waits own ds stores)

      // A fragment (elements 0-7: K=8g.., 8-15: K=16+8g..)
      union { v16h v; u32x4 q[2]; } fa;
      const h16* Ab = &As[c][0];
      const h16* Bb = &Bs[c][0];
      fa.q[0] = *(const u32x4*)(Ab + mrow * 40 + 8 * g);
      fa.q[1] = *(const u32x4*)(Ab + mrow * 40 + 16 + 8 * g);

      union { v16h v; u32x4 q[2]; } fb[4];
#pragma unroll
      for (int j = 0; j < 4; ++j) {
        const int co = nq * 64 + j * 16 + n15;
        fb[j].q[0] = *(const u32x4*)(Bb + co * 32 + 8 * g);
        fb[j].q[1] = *(const u32x4*)(Bb + co * 32 + 16 + 8 * g);
      }
#pragma unroll
      for (int j = 0; j < 4; ++j)
        acc[j] = __builtin_amdgcn_wmma_f32_16x16x32_f16(
            false, fa.v, false, fb[j].v, (short)0, acc[j], false, false);

      __syncthreads();   // all waves done reading buffer c
      if (ks + 2 < 72) { issueA(xh, ks + 2, c); issueB(wp, ks + 2, c); }
    }

    // BN + LeakyReLU for this branch, accumulate into sum
#pragma unroll
    for (int j = 0; j < 4; ++j) {
      const int co = nq * 64 + j * 16 + n15;
      const float scv = sc[br * 256 + co];
      const float shv = sh[br * 256 + co];
#pragma unroll
      for (int r = 0; r < 8; ++r) {
        float y = acc[j][r] * scv + shv;
        y = (y > 0.f) ? y : 0.01f * y;
        sum[j][r] += y;
      }
    }
  }

  __syncthreads();
#pragma unroll
  for (int j = 0; j < 4; ++j) {
    const int co = nq * 64 + j * 16 + n15;
#pragma unroll
    for (int r = 0; r < 8; ++r) {
      const int M = msub * 16 + g * 8 + r;
      Os[M * 256 + co] = sum[j][r];
    }
  }
  __syncthreads();
  for (int it = 0; it < 32; ++it) {
    const int co = it * 8 + (t >> 5);
    const int m = t & 31;
    out[((size_t)(b * C_ + co)) * HW_ + h * W_ + w0 + m] = Os[m * 256 + co];
  }
}

extern "C" void kernel_launch(void* const* d_in, const int* in_sizes, int n_in,
                              void* d_out, int out_size, void* d_ws, size_t ws_size,
                              hipStream_t stream) {
  (void)in_sizes; (void)n_in; (void)out_size; (void)ws_size;
  const float* f1 = (const float*)d_in[0];
  const float* f2 = (const float*)d_in[1];
  const float* w1r = (const float*)d_in[2];
  const float* b1r = (const float*)d_in[3];
  const float* w2r = (const float*)d_in[4];
  const float* b2r = (const float*)d_in[5];
  const float* w1d = (const float*)d_in[6];
  const float* b1d = (const float*)d_in[7];
  const float* w2d = (const float*)d_in[8];
  const float* b2d = (const float*)d_in[9];
  const float* qr_w = (const float*)d_in[10];
  const float* qr_b = (const float*)d_in[11];
  const float* kr_w = (const float*)d_in[12];
  const float* kr_b = (const float*)d_in[13];
  const float* vr_w = (const float*)d_in[14];
  const float* vr_b = (const float*)d_in[15];
  const float* qd_w = (const float*)d_in[16];
  const float* qd_b = (const float*)d_in[17];
  const float* kd_w = (const float*)d_in[18];
  const float* kd_b = (const float*)d_in[19];
  const float* vd_w = (const float*)d_in[20];
  const float* vd_b = (const float*)d_in[21];
  const float* convr_w = (const float*)d_in[22];
  const float* convr_b = (const float*)d_in[23];
  const float* bnr_g = (const float*)d_in[24];
  const float* bnr_b = (const float*)d_in[25];
  const float* bnr_m = (const float*)d_in[26];
  const float* bnr_v = (const float*)d_in[27];
  const float* convd_w = (const float*)d_in[28];
  const float* convd_b = (const float*)d_in[29];
  const float* bnd_g = (const float*)d_in[30];
  const float* bnd_b = (const float*)d_in[31];
  const float* bnd_m = (const float*)d_in[32];
  const float* bnd_v = (const float*)d_in[33];
  float* out = (float*)d_out;

  // workspace layout
  float* gr   = (float*)d_ws;          // 4096
  float* gd   = gr + 4096;             // 4096
  float* wrw  = gd + 4096;             // 4096
  float* wdw  = wrw + 4096;            // 4096
  float* f1in = wdw + 4096;            // 65536
  float* f2in = f1in + 65536;          // 65536
  float* s1   = f2in + 65536;          // 65536
  float* s2   = s1 + 65536;            // 65536
  float* sc   = s2 + 65536;            // 512
  float* sh   = sc + 512;              // 512
  h16* wp  = (h16*)(sh + 512);         // 2*589824 halves
  h16* xh1 = wp + (size_t)2 * 589824;  // 16777216 halves
  h16* xh2 = xh1 + (size_t)16777216;   // 16777216 halves

  k_stats<<<4096, 256, 0, stream>>>(f1, f2, gr, gd);
  k_pool<<<256, 256, 0, stream>>>(f1, f2, f1in, f2in);
  k_mlp<<<16, 256, 0, stream>>>(gr, gd, w1r, b1r, w2r, b2r, w1d, b1d, w2d, b2d,
                                wrw, wdw, out);
  k_attn<<<dim3(16, 16), 256, 0, stream>>>(f1in, f2in,
                                           qr_w, qr_b, kr_w, kr_b, vr_w, vr_b,
                                           qd_w, qd_b, kd_w, kd_b, vd_w, vd_b,
                                           s1, s2);
  k_bnprep<<<2, 256, 0, stream>>>(bnr_g, bnr_b, bnr_m, bnr_v, convr_b,
                                  bnd_g, bnd_b, bnd_m, bnd_v, convd_b, sc, sh);
  k_wpack<<<4608, 256, 0, stream>>>(convr_w, convd_w, wp);
  k_pack<<<dim3(64, 16), 256, 0, stream>>>(f1, f2, wrw, wdw, s1, s2, xh1, xh2);
  k_conv<<<dim3(128, 16), 256, 0, stream>>>(xh1, xh2, wp, wp + (size_t)589824,
                                            sc, sh, out);
}
